// QuantumResidualModel_56805237457114
// MI455X (gfx1250) — matrix operand order repaired and checked by reading, not compiled, not run on hardware
//
#include <hip/hip_runtime.h>
#include <math.h>

// ---------------- CDNA5 WMMA types ----------------
typedef __attribute__((ext_vector_type(16))) _Float16 v16h;
typedef __attribute__((ext_vector_type(8)))  float    v8f;

#define NQ 7
#define NAMP 128   // 2^7 amplitudes

// =====================================================================
// Kernel 1: 7-qubit statevector simulation, one sample per thread.
// State lives in LDS: 64 threads * 128 float2 = 64KB. Per-thread region
// is XOR-swizzled by (lane&31) so that same-logical-index accesses from
// the 32 lanes of a wave land on distinct bank pairs (conflict-free).
// =====================================================================
__global__ __launch_bounds__(64) void qsim_kernel(
    const float* __restrict__ x_q,   // [B,7]
    const float* __restrict__ pre_W, // [7,7]
    const float* __restrict__ pre_b, // [7]
    const float* __restrict__ q_ry,  // [3,7]
    const float* __restrict__ q_rz,  // [3,7]
    float* __restrict__ q_out,       // [B,28]
    int B)
{
  __shared__ float2 st[64 * NAMP];   // 65536 bytes
  const int tid = threadIdx.x;
  const int b   = blockIdx.x * 64 + tid;
  if (b >= B) return;

  const int sw = tid & 31;
  float2* s = &st[tid * NAMP];
  auto AMP = [&](int i) -> float2& { return s[i ^ sw]; };

  // angles = (tanh(x_q @ pre_W.T + pre_b) + 1) * pi/2
  float xq[NQ];
  #pragma unroll
  for (int k = 0; k < NQ; ++k) xq[k] = x_q[b * NQ + k];
  float ang[NQ];
  #pragma unroll
  for (int q = 0; q < NQ; ++q) {
    float acc = pre_b[q];
    #pragma unroll
    for (int k = 0; k < NQ; ++k) acc += xq[k] * pre_W[q * NQ + k];
    ang[q] = (tanhf(acc) + 1.0f) * 1.5707963267948966f;
  }

  // |0...0>
  for (int i = 0; i < NAMP; ++i) AMP(i) = make_float2(0.f, 0.f);
  AMP(0) = make_float2(1.f, 0.f);

  // qubit q has stride 2^(6-q) in the flat index (axis order of reference)
  auto apply_ry = [&](int q, float theta) {
    float c = cosf(0.5f * theta), sn = sinf(0.5f * theta);
    int str = 1 << (6 - q);
    for (int p = 0; p < 64; ++p) {
      int i0 = ((p & ~(str - 1)) << 1) | (p & (str - 1));
      int i1 = i0 | str;
      float2 a0 = AMP(i0), a1 = AMP(i1);
      AMP(i0) = make_float2(c * a0.x - sn * a1.x, c * a0.y - sn * a1.y);
      AMP(i1) = make_float2(sn * a0.x + c * a1.x, sn * a0.y + c * a1.y);
    }
  };
  auto apply_rz = [&](int q, float theta) {
    float c = cosf(0.5f * theta), sn = sinf(0.5f * theta);
    int str = 1 << (6 - q);
    for (int i = 0; i < NAMP; ++i) {
      float2 a = AMP(i);
      float sg = (i & str) ? sn : -sn;   // bit0: e^{-i t/2}; bit1: e^{+i t/2}
      AMP(i) = make_float2(a.x * c - a.y * sg, a.y * c + a.x * sg);
    }
  };

  for (int layer = 0; layer < 3; ++layer) {
    for (int q = 0; q < NQ; ++q) apply_ry(q, ang[q]);                // data re-upload
    for (int q = 0; q < NQ; ++q) {                                   // trainable
      apply_ry(q, q_ry[layer * NQ + q]);
      apply_rz(q, q_rz[layer * NQ + q]);
    }
    int shift = layer + 1;
    for (int q = 0; q < NQ; ++q) {                                   // CNOT ring
      int t = (q + shift) % NQ;
      int sc = 1 << (6 - q), stt = 1 << (6 - t);
      for (int i = 0; i < NAMP; ++i) {
        if ((i & sc) && !(i & stt)) {
          float2 a = AMP(i), bb = AMP(i | stt);
          AMP(i) = bb; AMP(i | stt) = a;
        }
      }
    }
  }

  // expectations: Z(7), X(7), Y(7), ZZ(7)
  float z[NQ], zz[NQ];
  #pragma unroll
  for (int q = 0; q < NQ; ++q) { z[q] = 0.f; zz[q] = 0.f; }
  for (int i = 0; i < NAMP; ++i) {
    float2 a = AMP(i);
    float p = a.x * a.x + a.y * a.y;
    #pragma unroll
    for (int q = 0; q < NQ; ++q) {
      float sq = ((i >> (6 - q)) & 1) ? -1.f : 1.f;
      int j = (q + 1) % NQ;
      float sj = ((i >> (6 - j)) & 1) ? -1.f : 1.f;
      z[q]  += sq * p;
      zz[q] += sq * sj * p;
    }
  }
  float* o = q_out + (size_t)b * 28;
  #pragma unroll
  for (int q = 0; q < NQ; ++q) { o[q] = z[q]; o[21 + q] = zz[q]; }
  for (int q = 0; q < NQ; ++q) {
    int str = 1 << (6 - q);
    float ar = 0.f, ai = 0.f;
    for (int p = 0; p < 64; ++p) {
      int i0 = ((p & ~(str - 1)) << 1) | (p & (str - 1));
      float2 a0 = AMP(i0), a1 = AMP(i0 | str);
      ar += a0.x * a1.x + a0.y * a1.y;   // Re conj(a0)*a1
      ai += a0.x * a1.y - a0.y * a1.x;   // Im conj(a0)*a1
    }
    o[7 + q]  = 2.f * ar;  // <X>
    o[14 + q] = 2.f * ai;  // <Y>
  }
}

// =====================================================================
// Kernel 2: fused MLP via v_wmma_f32_16x16x32_f16. One wave = one
// 16-sample tile; activations staged in per-wave LDS ([16][72] halves).
// =====================================================================
#define STW 72   // LDS row stride (halves), padded

// A operand, 16-bit 16x32 layout (ISA 7.12.2):
//   lane l: M = l&15 ; half h -> K = (h<8 ? h : h+8) + 8*(l>=16)
__device__ inline v16h load_A_global(const float* X, int ld, int m0, int k0, int lane) {
  v16h a;
  int m = m0 + (lane & 15);
  int kb = k0 + ((lane >> 4) << 3);
  #pragma unroll
  for (int h = 0; h < 16; ++h) {
    int k = kb + (h < 8 ? h : h + 8);
    a[h] = (_Float16)X[(size_t)m * ld + k];
  }
  return a;
}

__device__ inline v16h load_A_stage(const _Float16* st, int k0, int lane) {
  v16h a;
  int m = lane & 15;
  int kb = k0 + ((lane >> 4) << 3);
  #pragma unroll
  for (int h = 0; h < 16; ++h) {
    int k = kb + (h < 8 ? h : h + 8);
    a[h] = st[m * STW + k];
  }
  return a;
}

// B operand, 32x16 16-bit layout: lane l: N = l&15 ; half h -> K = (h<8 ? h : h+8) + 8*(l>=16)
// Weights W are [Nreal, Kreal] row-major f32; out = act @ W^T so B[k][n] = W[n][k].
__device__ inline v16h load_B_w(const float* W, int Kreal, int n0, int k0, int lane) {
  v16h bv;
  int n = n0 + (lane & 15);
  int kb = k0 + ((lane >> 4) << 3);
  #pragma unroll
  for (int h = 0; h < 16; ++h) {
    int k = kb + (h < 8 ? h : h + 8);
    float w = (k < Kreal) ? W[(size_t)n * Kreal + k] : 0.0f;
    bv[h] = (_Float16)w;
  }
  return bv;
}

__device__ inline v8f wmma_f16(v16h a, v16h b, v8f c) {
  return __builtin_amdgcn_wmma_f32_16x16x32_f16(false, a, false, b, (short)0, c, false, false);
}

// C/D layout: lane l, reg r: M = r + 8*(l>=16), N = l&15. Fuse bias + ReLU.
__device__ inline void store_C_relu(_Float16* st, v8f acc, const float* bias,
                                    int nb, int cb, int lane) {
  int nl = lane & 15;
  float b = bias[nb + nl];
  int mB = (lane >> 4) << 3;
  #pragma unroll
  for (int r = 0; r < 8; ++r) {
    float v = acc[r] + b;
    st[(mB + r) * STW + cb + nl] = (_Float16)(v > 0.f ? v : 0.f);
  }
}

__global__ __launch_bounds__(128) void mlp_kernel(
    const float* __restrict__ x_c,    // [B,64]
    const float* __restrict__ q_out,  // [B,28]
    const float* __restrict__ eW1, const float* __restrict__ eb1,   // [32,64],[32]
    const float* __restrict__ eW2, const float* __restrict__ eb2,   // [16,32],[16]
    const float* __restrict__ fW1, const float* __restrict__ fb1,   // [64,44],[64]
    const float* __restrict__ fW2, const float* __restrict__ fb2,   // [32,64],[32]
    const float* __restrict__ fW3, const float* __restrict__ fb3,   // [1,32],[1]
    const float* __restrict__ corr_scale,
    float* __restrict__ out, int B)
{
  __shared__ _Float16 stage_all[4][16 * STW];
  const int lane = threadIdx.x & 31;
  const int w    = threadIdx.x >> 5;
  const int tile = blockIdx.x * 4 + w;
  const int m0   = tile * 16;
  _Float16* st = stage_all[w];

  // ---- L1: relu(x_c @ eW1^T + eb1) -> stage cols 0..31 ----
  v16h a0 = load_A_global(x_c, 64, m0, 0, lane);
  v16h a1 = load_A_global(x_c, 64, m0, 32, lane);
  #pragma unroll
  for (int nt = 0; nt < 2; ++nt) {
    v8f acc = {};
    acc = wmma_f16(a0, load_B_w(eW1, 64, nt * 16, 0,  lane), acc);
    acc = wmma_f16(a1, load_B_w(eW1, 64, nt * 16, 32, lane), acc);
    store_C_relu(st, acc, eb1, nt * 16, nt * 16, lane);
  }
  __syncthreads();

  // ---- L2: relu(L1 @ eW2^T + eb2) -> stage cols 28..43 (concat slot) ----
  {
    v16h a = load_A_stage(st, 0, lane);
    v8f acc = {};
    acc = wmma_f16(a, load_B_w(eW2, 32, 0, 0, lane), acc);
    __syncthreads();
    store_C_relu(st, acc, eb2, 0, 28, lane);
  }
  // q_out -> cols 0..27 ; zero-pad cols 44..63
  for (int idx = lane; idx < 16 * 28; idx += 32) {
    int m = idx / 28, k = idx % 28;
    st[m * STW + k] = (_Float16)q_out[(size_t)(m0 + m) * 28 + k];
  }
  for (int idx = lane; idx < 16 * 20; idx += 32) {
    int m = idx / 20, k = 44 + idx % 20;
    st[m * STW + k] = (_Float16)0.f;
  }
  __syncthreads();

  // ---- L3: relu(h[16,44->64] @ fW1^T + fb1) -> [16,64] ----
  a0 = load_A_stage(st, 0, lane);
  a1 = load_A_stage(st, 32, lane);
  __syncthreads();
  #pragma unroll
  for (int nt = 0; nt < 4; ++nt) {
    v8f acc = {};
    acc = wmma_f16(a0, load_B_w(fW1, 44, nt * 16, 0,  lane), acc);
    acc = wmma_f16(a1, load_B_w(fW1, 44, nt * 16, 32, lane), acc);  // k>=44 zero-padded
    store_C_relu(st, acc, fb1, nt * 16, nt * 16, lane);
  }
  __syncthreads();

  // ---- L4: relu(L3 @ fW2^T + fb2) -> [16,32] ----
  a0 = load_A_stage(st, 0, lane);
  a1 = load_A_stage(st, 32, lane);
  __syncthreads();
  #pragma unroll
  for (int nt = 0; nt < 2; ++nt) {
    v8f acc = {};
    acc = wmma_f16(a0, load_B_w(fW2, 64, nt * 16, 0,  lane), acc);
    acc = wmma_f16(a1, load_B_w(fW2, 64, nt * 16, 32, lane), acc);
    store_C_relu(st, acc, fb2, nt * 16, nt * 16, lane);
  }
  __syncthreads();

  // ---- L5: (L4 @ fW3^T + fb3) * corr_scale -> out[m0..m0+15] ----
  if (lane < 16) {
    float sum = fb3[0];
    #pragma unroll
    for (int k = 0; k < 32; ++k) sum += (float)st[lane * STW + k] * fW3[k];
    out[m0 + lane] = corr_scale[0] * sum;
  }
}

// =====================================================================
extern "C" void kernel_launch(void* const* d_in, const int* in_sizes, int n_in,
                              void* d_out, int out_size, void* d_ws, size_t ws_size,
                              hipStream_t stream) {
  const float* x_q    = (const float*)d_in[0];
  const float* x_c    = (const float*)d_in[1];
  const float* pre_W  = (const float*)d_in[2];
  const float* pre_b  = (const float*)d_in[3];
  const float* q_ry   = (const float*)d_in[4];
  const float* q_rz   = (const float*)d_in[5];
  const float* eW1    = (const float*)d_in[6];
  const float* eb1    = (const float*)d_in[7];
  const float* eW2    = (const float*)d_in[8];
  const float* eb2    = (const float*)d_in[9];
  const float* fW1    = (const float*)d_in[10];
  const float* fb1    = (const float*)d_in[11];
  const float* fW2    = (const float*)d_in[12];
  const float* fb2    = (const float*)d_in[13];
  const float* fW3    = (const float*)d_in[14];
  const float* fb3    = (const float*)d_in[15];
  const float* cscale = (const float*)d_in[16];

  const int B = in_sizes[0] / NQ;      // 16384
  float* q_expect = (float*)d_ws;      // [B,28] scratch

  qsim_kernel<<<B / 64, 64, 0, stream>>>(x_q, pre_W, pre_b, q_ry, q_rz, q_expect, B);
  mlp_kernel<<<B / 64, 128, 0, stream>>>(x_c, q_expect, eW1, eb1, eW2, eb2,
                                         fW1, fb1, fW2, fb2, fW3, fb3,
                                         cscale, (float*)d_out, B);
}
